// CEBlock_71210557767850
// MI455X (gfx1250) — compile-verified, matrix operand order checked by reading, not run on hardware
//
#include <hip/hip_runtime.h>
#include <hip/hip_bf16.h>
#include <stdint.h>

typedef __bf16 bf16_t;
typedef bf16_t v16bf __attribute__((ext_vector_type(16)));
typedef bf16_t v8bf  __attribute__((ext_vector_type(8)));
typedef float  v8f   __attribute__((ext_vector_type(8)));

#define B_   32
#define H_   12
#define N_   320
#define T_   64
#define S_   256
#define C_   768
#define HH_  3072
#define KEEP_ 179
#define REM_  77
#define BN_  (B_*N_)   // 10240

union QK { unsigned long long u64; bf16_t h[4]; };

// ---------------------------------------------------------------- LayerNorm
__global__ void ln_kernel(const float* __restrict__ in, const float* __restrict__ sc,
                          const float* __restrict__ bi, float* __restrict__ outF,
                          bf16_t* __restrict__ outH, int addInput)
{
  __shared__ float r1[256], r2[256];
  int row = blockIdx.x;
  const float* p = in + (size_t)row * C_;
  float v[3]; float s = 0.f, ss = 0.f;
#pragma unroll
  for (int i = 0; i < 3; i++) {
    int c = threadIdx.x + i * 256;
    v[i] = p[c]; s += v[i]; ss += v[i] * v[i];
  }
  r1[threadIdx.x] = s; r2[threadIdx.x] = ss; __syncthreads();
  for (int st = 128; st > 0; st >>= 1) {
    if (threadIdx.x < st) { r1[threadIdx.x] += r1[threadIdx.x + st]; r2[threadIdx.x] += r2[threadIdx.x + st]; }
    __syncthreads();
  }
  float mean = r1[0] * (1.0f / C_);
  float var  = r2[0] * (1.0f / C_) - mean * mean;
  float inv  = rsqrtf(var + 1e-5f);
#pragma unroll
  for (int i = 0; i < 3; i++) {
    int c = threadIdx.x + i * 256;
    float o = (v[i] - mean) * inv * sc[c] + bi[c];
    if (addInput) o += v[i];
    if (outF) outF[(size_t)row * C_ + c] = o;
    if (outH) outH[(size_t)row * C_ + c] = (bf16_t)o;
  }
}

// ---------------------------------------------------------------- WMMA GEMM
// BM=64, BN=128, BK=32. 8 wave32 waves in 2(M) x 4(N), each wave a 32x32 tile
// via 2x2 v_wmma_f32_16x16x32_bf16. Compile-time specialized.
// LDS: padded (80B row stride) fragment-ready layouts -> ds_load_b128 frags,
// quad staging -> global_load_b64/b128 + ds_store_b64.
// Preconditions (guaranteed by launches): M % 64 == 0, K % 32 == 0,
// all element strides/offsets multiples of 4 (for 8/16B alignment),
// and if N_FULL then Nc % 128 == 0.
template <bool A_BF16, bool B_BF16, bool B_TRANS, int ACT,
          bool HAS_BIAS, bool HAS_RES, bool OUT_F, bool OUT_H, bool N_FULL>
__global__ __launch_bounds__(256) void wmma_gemm(
    const void* __restrict__ Ap, long long aOuter, long long aInner, int lda,
    const void* __restrict__ Bp, long long bOuter, long long bInner, int ldb,
    const float* __restrict__ bias, const float* __restrict__ res,
    float* __restrict__ outF, bf16_t* __restrict__ outH,
    long long cOuter, long long cInner, int ldc,
    int M, int Nc, int K, int batchInner)
{
  // 80B row stride: rows 16B-aligned; the 16 lanes' 16B fragment windows
  // (stride 20 banks) tile all 64 LDS banks disjointly -> conflict-free.
  __shared__ __align__(16) bf16_t As[64][40];
  __shared__ __align__(16) bf16_t Bs[128][40];   // column-major B tile

  int z   = blockIdx.z;
  int bi_ = z / batchInner, hi_ = z % batchInner;
  long long aOff = (long long)bi_ * aOuter + (long long)hi_ * aInner;
  long long bOff = (long long)bi_ * bOuter + (long long)hi_ * bInner;
  long long cOff = (long long)bi_ * cOuter + (long long)hi_ * cInner;

  int tid  = threadIdx.x;
  int lane = tid & 31, wid = tid >> 5;
  int wm = wid & 1, wn = wid >> 1;
  int blockM = blockIdx.y * 64;
  int blockN = blockIdx.x * 128;

  v8f acc[2][2] = {};

  const float*  Af = (const float*)Ap;
  const bf16_t* Ah = (const bf16_t*)Ap;
  const float*  Bf = (const float*)Bp;
  const bf16_t* Bh = (const bf16_t*)Bp;

  for (int k0 = 0; k0 < K; k0 += 32) {
    // ---- stage A (64x32): 2 quads per thread; f32 path uses float4 loads
#pragma unroll
    for (int e = 0; e < 2; e++) {
      int q = tid + e * 256;             // 0..511 quads
      int r = q >> 3;                    // 0..63
      int c = (q & 7) * 4;               // 0,4,...,28
      long long a = aOff + (long long)(blockM + r) * lda + (k0 + c);
      QK pk;
      if (A_BF16) {
        pk.u64 = *(const unsigned long long*)(Ah + a);
      } else {
        float4 f = *(const float4*)(Af + a);
        pk.h[0] = (bf16_t)f.x; pk.h[1] = (bf16_t)f.y;
        pk.h[2] = (bf16_t)f.z; pk.h[3] = (bf16_t)f.w;
      }
      *(unsigned long long*)&As[r][c] = pk.u64;
    }
    // ---- stage B (32x128 logical, stored column-major): 4 quads per thread
#pragma unroll
    for (int e = 0; e < 4; e++) {
      int q  = tid + e * 256;            // 0..1023 quads
      int c  = q & 127;
      int kk = (q >> 7) * 4;             // 0,4,...,28
      int gc = blockN + c, gk = k0 + kk;
      QK pk; pk.u64 = 0ull;
      if (N_FULL || gc < Nc) {
        if (B_TRANS) {
          long long a = bOff + (long long)gc * ldb + gk;   // contiguous quad
          if (B_BF16) {
            pk.u64 = *(const unsigned long long*)(Bh + a);
          } else {
            float4 f = *(const float4*)(Bf + a);
            pk.h[0] = (bf16_t)f.x; pk.h[1] = (bf16_t)f.y;
            pk.h[2] = (bf16_t)f.z; pk.h[3] = (bf16_t)f.w;
          }
        } else {
          long long a = bOff + (long long)gk * ldb + gc;   // strided by ldb
#pragma unroll
          for (int i = 0; i < 4; i++) {
            pk.h[i] = B_BF16 ? Bh[a] : (bf16_t)Bf[a];
            a += ldb;
          }
        }
      }
      *(unsigned long long*)&Bs[c][kk] = pk.u64;
    }
    // prefetch next K tile into cache while WMMAs run (global_prefetch_b8)
    if (k0 + 32 < K) {
      long long pa = aOff + (long long)(blockM + (tid >> 2)) * lda + (k0 + 32) + (tid & 3) * 8;
      __builtin_prefetch((const char*)Ap + pa * (A_BF16 ? 2 : 4), 0, 1);
      long long pb = B_TRANS
          ? bOff + (long long)(blockN + (tid >> 1)) * ldb + (k0 + 32) + (tid & 1) * 16
          : bOff + (long long)(k0 + 32 + (tid >> 4)) * ldb + blockN + (tid & 15) * 8;
      __builtin_prefetch((const char*)Bp + pb * (B_BF16 ? 2 : 4), 0, 1);
    }
    __syncthreads();

    // ---- fragments: lane<16 -> K 0..7 & 16..23, lane>=16 -> K 8..15 & 24..31
    int lr = lane & 15;
    int kb = (lane >> 4) * 8;
    v16bf afr[2], bfr[2];
#pragma unroll
    for (int i = 0; i < 2; i++) {
      int r = wm * 32 + i * 16 + lr;
      v8bf lo = *(const v8bf*)&As[r][kb];
      v8bf hi = *(const v8bf*)&As[r][16 + kb];
      afr[i] = __builtin_shufflevector(lo, hi, 0,1,2,3,4,5,6,7,8,9,10,11,12,13,14,15);
    }
#pragma unroll
    for (int i = 0; i < 2; i++) {
      int c = wn * 32 + i * 16 + lr;
      v8bf lo = *(const v8bf*)&Bs[c][kb];
      v8bf hi = *(const v8bf*)&Bs[c][16 + kb];
      bfr[i] = __builtin_shufflevector(lo, hi, 0,1,2,3,4,5,6,7,8,9,10,11,12,13,14,15);
    }
#pragma unroll
    for (int i = 0; i < 2; i++)
#pragma unroll
      for (int j = 0; j < 2; j++)
        acc[i][j] = __builtin_amdgcn_wmma_f32_16x16x32_bf16(
            false, afr[i], false, bfr[j], (short)0, acc[i][j], false, false);
    __syncthreads();
  }

  // ---- epilogue: elem e -> row = e + (lane>=16 ? 8:0), col = lane&15
  int lr    = lane & 15;
  int rbase = (lane >> 4) * 8;
#pragma unroll
  for (int i = 0; i < 2; i++) {
#pragma unroll
    for (int j = 0; j < 2; j++) {
      int col = blockN + wn * 32 + j * 16 + lr;
      if (!N_FULL && col >= Nc) continue;
      float bv = HAS_BIAS ? bias[col] : 0.f;
#pragma unroll
      for (int e = 0; e < 8; e++) {
        int row = blockM + wm * 32 + i * 16 + rbase + e;   // row < M (M%64==0)
        float o = acc[i][j][e];
        if (HAS_BIAS) o += bv;
        if (ACT == 1) o = 0.5f * o * (1.0f + erff(o * 0.70710678118f)); // exact GELU
        long long ci = cOff + (long long)row * ldc + col;
        if (HAS_RES) o += res[ci];
        if (OUT_F)   outF[ci] = o;
        if (OUT_H)   outH[ci] = (bf16_t)o;
      }
    }
  }
}

// ---------------------------------------------------------------- attention softmax
__global__ void attn_softmax(float* __restrict__ a, float scale)
{
  __shared__ float red[320];
  float* p = a + (size_t)blockIdx.x * N_;
  int t = threadIdx.x;
  float v = p[t] * scale;
  red[t] = v; __syncthreads();
  for (int st = 256; st > 0; st >>= 1) { if (t < st && t + st < N_) red[t] = fmaxf(red[t], red[t + st]); __syncthreads(); }
  float m = red[0]; __syncthreads();
  float e = __expf(v - m);
  red[t] = e; __syncthreads();
  for (int st = 256; st > 0; st >>= 1) { if (t < st && t + st < N_) red[t] += red[t + st]; __syncthreads(); }
  p[t] = e / red[0];
}

// ---------------------------------------------------------------- w_ts + score_s
__global__ void wts_kernel(const float* __restrict__ attn, float* __restrict__ score,
                           bf16_t* __restrict__ wts, bf16_t* __restrict__ wtsT)
{
  __shared__ float red[256];
  int b = blockIdx.x >> 6;
  int t = blockIdx.x & 63;
  int s = threadIdx.x;
  float acc = 0.f;
#pragma unroll
  for (int h = 0; h < H_; h++)
    acc += attn[((size_t)(b * H_ + h) * N_ + t) * N_ + T_ + s];
  float raw = acc * (1.0f / H_);
  atomicAdd(&score[b * S_ + s], raw * (1.0f / T_));
  red[s] = raw; __syncthreads();
  for (int st = 128; st > 0; st >>= 1) { if (s < st) red[s] = fmaxf(red[s], red[s + st]); __syncthreads(); }
  float m = red[0]; __syncthreads();
  float e = __expf(raw - m);
  red[s] = e; __syncthreads();
  for (int st = 128; st > 0; st >>= 1) { if (s < st) red[s] += red[s + st]; __syncthreads(); }
  float w = e / red[0];
  wts [((size_t)(b * T_ + t)) * S_ + s] = (bf16_t)w;
  wtsT[((size_t)(b * S_ + s)) * T_ + t] = (bf16_t)w;
}

__global__ void zero_f32(float* p, int n)
{
  int i = blockIdx.x * blockDim.x + threadIdx.x;
  if (i < n) p[i] = 0.f;
}

// ---------------------------------------------------------------- sort + select
__global__ void sort_select(const float* __restrict__ score,
                            const int* __restrict__ git, const int* __restrict__ gis,
                            float* __restrict__ gitOut, float* __restrict__ gsOut,
                            float* __restrict__ remOut, int* __restrict__ idxbuf)
{
  __shared__ float ss[256];
  __shared__ int   si[256];
  int b = blockIdx.x, t = threadIdx.x;
  ss[t] = score[b * S_ + t]; si[t] = t; __syncthreads();
  for (int k = 2; k <= 256; k <<= 1) {
    for (int j = k >> 1; j > 0; j >>= 1) {
      int ixj = t ^ j;
      if (ixj > t) {
        float s1 = ss[t], s2 = ss[ixj]; int i1 = si[t], i2 = si[ixj];
        bool firstGreater = (s1 > s2) || (s1 == s2 && i1 < i2);
        bool desc = ((t & k) == 0);
        if (desc ? !firstGreater : firstGreater) {
          ss[t] = s2; ss[ixj] = s1; si[t] = i2; si[ixj] = i1;
        }
      }
      __syncthreads();
    }
  }
  if (t < KEEP_) {
    int keep = si[t];
    gsOut[b * KEEP_ + t] = (float)gis[b * S_ + keep];
    idxbuf[b * (T_ + KEEP_) + T_ + t] = keep + T_;
  } else {
    remOut[b * REM_ + (t - KEEP_)] = (float)si[t];
  }
  if (t < T_) {
    int g = git[b * T_ + t];
    gitOut[b * T_ + t] = (float)g;
    idxbuf[b * (T_ + KEEP_) + t] = g;
  }
}

// ---------------------------------------------------------------- final gather
__global__ void gather_rows(const float* __restrict__ xfin, const int* __restrict__ idxbuf,
                            float* __restrict__ xout)
{
  int j = blockIdx.x;                 // 0 .. B*243-1
  int b = j / (T_ + KEEP_);
  int r = idxbuf[j];
  const float* src = xfin + ((size_t)b * N_ + r) * C_;
  float* dst = xout + (size_t)j * C_;
#pragma unroll
  for (int i = 0; i < 3; i++) { int c = threadIdx.x + i * 256; dst[c] = src[c]; }
}

// ================================================================ host
extern "C" void kernel_launch(void* const* d_in, const int* in_sizes, int n_in,
                              void* d_out, int out_size, void* d_ws, size_t ws_size,
                              hipStream_t stream)
{
  (void)in_sizes; (void)n_in; (void)out_size; (void)ws_size;
  const float* x      = (const float*)d_in[0];
  const int*   git    = (const int*)  d_in[1];
  const int*   gis    = (const int*)  d_in[2];
  const float* n1s    = (const float*)d_in[3];
  const float* n1b    = (const float*)d_in[4];
  const float* qkv_w  = (const float*)d_in[5];
  const float* qkv_b  = (const float*)d_in[6];
  const float* proj_w = (const float*)d_in[7];
  const float* proj_b = (const float*)d_in[8];
  const float* n2s    = (const float*)d_in[9];
  const float* n2b    = (const float*)d_in[10];
  const float* fc1_w  = (const float*)d_in[11];
  const float* fc1_b  = (const float*)d_in[12];
  const float* fc2_w  = (const float*)d_in[13];
  const float* fc2_b  = (const float*)d_in[14];
  const float* gcn_w  = (const float*)d_in[15];
  const float* gcn_b  = (const float*)d_in[16];
  const float* n3s    = (const float*)d_in[17];
  const float* n3b    = (const float*)d_in[18];

  // workspace layout (lifetime-based reuse, peak ~144MB)
  char* ws = (char*)d_ws;
  const size_t MB = 1ull << 20;
  bf16_t* h1    = (bf16_t*)(ws +   0 * MB);  // LN1(x) bf16, dead after qkv
  bf16_t* qkvb  = (bf16_t*)(ws +  16 * MB);  // fused qkv bf16 (10240x2304)
  bf16_t* xa    = (bf16_t*)(ws +   0 * MB);  // attn@V bf16 (reuses h1)
  float*  x1    = (float*) (ws +  64 * MB);  // after proj + residual
  bf16_t* h2    = (bf16_t*)(ws +  96 * MB);  // LN2(x1) bf16
  bf16_t* tbuf  = (bf16_t*)(ws +   0 * MB);  // GELU(fc1) bf16 (10240x3072)
  float*  x2f   = (float*) (ws + 112 * MB);  // after fc2 + residual
  bf16_t* x2h   = (bf16_t*)(ws +  96 * MB);  // bf16 copy (reuses h2)
  bf16_t* sup   = (bf16_t*)(ws +   0 * MB);  // support bf16 (reuses tbuf)
  bf16_t* wts   = (bf16_t*)(ws +  16 * MB);
  bf16_t* wtsT  = (bf16_t*)(ws +  17 * MB);
  float*  score = (float*) (ws +  18 * MB);
  int*    idxb  = (int*)   (ws +  19 * MB);
  float*  xg    = (float*) (ws +  32 * MB);  // concat(out_t, out_s)
  float*  xfin  = (float*) (ws +  64 * MB);  // xg + LN3(xg) (reuses x1)

  // d_out: x(32*243*768), git(32*64), gs(32*179), rem(32*77), attn(32*12*320*320)
  float* out    = (float*)d_out;
  float* xOut   = out;
  float* gitOut = out + (size_t)B_ * (T_ + KEEP_) * C_;
  float* gsOut  = gitOut + B_ * T_;
  float* remOut = gsOut + B_ * KEEP_;
  float* attn   = remOut + B_ * REM_;

  const long long QKV_BS = (long long)N_ * 3 * C_;   // 737280
  const long long ATT_H  = (long long)N_ * N_;       // 102400
  const long long ATT_B  = H_ * ATT_H;               // 1228800
  const long long XC_B   = (long long)N_ * C_;       // 245760

  // 1. h1 = LN1(x) -> bf16
  ln_kernel<<<BN_, 256, 0, stream>>>(x, n1s, n1b, nullptr, h1, 0);

  // 2. qkv = h1 @ qkv_w + qkv_b -> bf16
  wmma_gemm<true,false,false,0,true,false,false,true,true><<<dim3(18,160,1),256,0,stream>>>(
      h1, 0, 0, C_,  qkv_w, 0, 0, 3*C_,
      qkv_b, nullptr, nullptr, qkvb, 0, 0, 3*C_,
      BN_, 3*C_, C_, 1);

  // 3. attn logits = q @ k^T (batched over 384 (b,h), read in-place from qkv)
  wmma_gemm<true,true,true,0,false,false,true,false,false><<<dim3(3,5,B_*H_),256,0,stream>>>(
      qkvb,      QKV_BS, 64, 3*C_,
      qkvb + C_, QKV_BS, 64, 3*C_,
      nullptr, nullptr, attn, nullptr, ATT_B, ATT_H, N_,
      N_, N_, 64, H_);

  // 4. softmax rows (scale 1/sqrt(64) folded)
  attn_softmax<<<B_*H_*N_, 320, 0, stream>>>(attn, 0.125f);

  // 5. score_s accumulation + w_ts (+ transpose)
  zero_f32<<<(B_*S_ + 255)/256, 256, 0, stream>>>(score, B_*S_);
  wts_kernel<<<B_*T_, 256, 0, stream>>>(attn, score, wts, wtsT);

  // 6. xa = attn @ v -> bf16 into (B,N,C) with head offset
  wmma_gemm<false,true,false,0,false,false,false,true,false><<<dim3(1,5,B_*H_),256,0,stream>>>(
      attn,         ATT_B, ATT_H, N_,
      qkvb + 2*C_,  QKV_BS, 64, 3*C_,
      nullptr, nullptr, nullptr, xa, XC_B, 64, C_,
      N_, 64, N_, H_);

  // 7. x1 = xa @ proj_w + proj_b + x
  wmma_gemm<true,false,false,0,true,true,true,false,true><<<dim3(6,160,1),256,0,stream>>>(
      xa, 0, 0, C_,  proj_w, 0, 0, C_,
      proj_b, x, x1, nullptr, 0, 0, C_,
      BN_, C_, C_, 1);

  // 8. h2 = LN2(x1)
  ln_kernel<<<BN_, 256, 0, stream>>>(x1, n2s, n2b, nullptr, h2, 0);

  // 9. t = GELU(h2 @ fc1_w + fc1_b) -> bf16
  wmma_gemm<true,false,false,1,true,false,false,true,true><<<dim3(24,160,1),256,0,stream>>>(
      h2, 0, 0, C_,  fc1_w, 0, 0, HH_,
      fc1_b, nullptr, nullptr, tbuf, 0, 0, HH_,
      BN_, HH_, C_, 1);

  // 10. x2 = t @ fc2_w + fc2_b + x1 (f32 + bf16)
  wmma_gemm<true,false,false,0,true,true,true,true,true><<<dim3(6,160,1),256,0,stream>>>(
      tbuf, 0, 0, HH_,  fc2_w, 0, 0, C_,
      fc2_b, x1, x2f, x2h, 0, 0, C_,
      BN_, C_, HH_, 1);

  // 11. support = x2 @ gcn_w + gcn_b -> bf16
  wmma_gemm<true,false,false,0,true,false,false,true,true><<<dim3(6,160,1),256,0,stream>>>(
      x2h, 0, 0, C_,  gcn_w, 0, 0, C_,
      gcn_b, nullptr, nullptr, sup, 0, 0, C_,
      BN_, C_, C_, 1);

  // 12. out_t = w_ts @ support[:,64:] (batched over b)
  wmma_gemm<true,true,false,0,false,false,true,false,true><<<dim3(6,1,B_),256,0,stream>>>(
      wts,           (long long)T_*S_, 0, S_,
      sup + T_*C_,   XC_B, 0, C_,
      nullptr, nullptr, xg, nullptr, XC_B, 0, C_,
      T_, C_, S_, 1);

  // 13. out_s = w_ts^T @ support[:,:64]
  wmma_gemm<true,true,false,0,false,false,true,false,true><<<dim3(6,4,B_),256,0,stream>>>(
      wtsT,          (long long)S_*T_, 0, T_,
      sup,           XC_B, 0, C_,
      nullptr, nullptr, xg + T_*C_, nullptr, XC_B, 0, C_,
      S_, C_, T_, 1);

  // 14. xfin = xg + LN3(xg)
  ln_kernel<<<B_*N_, 256, 0, stream>>>(xg, n3s, n3b, xfin, nullptr, 1);

  // 15. sort scores, emit index outputs + gather index buffer
  sort_select<<<B_, 256, 0, stream>>>(score, git, gis, gitOut, gsOut, remOut, idxb);

  // 16. gather selected rows into x output
  gather_rows<<<B_*(T_+KEEP_), 256, 0, stream>>>(xfin, idxb, xOut);
}
